// BackupOpticalConvolution_28999619182923
// MI455X (gfx1250) — matrix operand order repaired and checked by reading, not compiled, not run on hardware
//
#include <hip/hip_runtime.h>
#include <hip/hip_bf16.h>

// Implicit-GEMM 3x3 conv (stride 1, pad 1) for MI455X (gfx1250, wave32).
//   Out[co, n] = sum_k W[co, k] * Xcol[k, n],  K = Cin*9 = 1152, N = B*H*W = 100352
// f32 inputs converted to f16 when staged into LDS; f32 WMMA accumulation.
// Double-buffered LDS; packed (2-wide) staging stores; one barrier per stage.

typedef __attribute__((ext_vector_type(16))) _Float16 v16h;
typedef __attribute__((ext_vector_type(8)))  _Float16 v8h;
typedef __attribute__((ext_vector_type(2)))  _Float16 h2;
typedef __attribute__((ext_vector_type(8)))  float    v8f;

#define CIN    128
#define COUT   256
#define HW     3136      // 56*56
#define KTOT   1152      // 128*9
#define NTOT   100352    // 32*3136
#define KB     64        // k per stage == 2 WMMA depths
#define NSTAGE (KTOT / KB)   // 18, exact
#define MT     128       // Cout tile per workgroup
#define NT     128       // pixel tile per workgroup
#define SROW   72        // LDS row stride in halves (64 + pad), 144B => 16B aligned rows

__global__ __launch_bounds__(256, 2)
void conv3x3_wmma_f16(const float* __restrict__ X,
                      const float* __restrict__ Wt,
                      const float* __restrict__ bias,
                      float* __restrict__ out) {
    __shared__ _Float16 As[2][MT * SROW];   // A tile: [m][k], m = local Cout
    __shared__ _Float16 Bs[2][NT * SROW];   // B tile transposed: [n][k], n = local pixel

    const int tid  = threadIdx.x;
    const int lane = tid & 31;
    const int wave = tid >> 5;          // 8 waves
    const int wm   = wave & 1;          // 2 wave rows  (64 Cout each)
    const int wn   = wave >> 1;         // 4 wave cols  (32 pixels each)
    const int half = lane >> 4;         // lane group 0/1
    const int l16  = lane & 15;

    const int m0 = blockIdx.y * MT;     // Cout tile base
    const int n0 = blockIdx.x * NT;     // pixel tile base

    // ---- staging helper (global f32 -> f16 -> LDS), fully inlined ----
    auto stage = [&](int buf, int ks) {
        const int k0 = ks * KB;
        // A: weights[m0+m][k0+k2..k2+1] -> As[buf][m][k2..k2+1]
        // consecutive tid -> consecutive k pairs (coalesced b64 loads)
        #pragma unroll
        for (int i = 0; i < (MT * KB) / (256 * 2); ++i) {
            const int idx = tid + i * 256;
            const int m  = idx >> 5;
            const int k2 = (idx & 31) << 1;
            const float2 wv = *(const float2*)&Wt[(m0 + m) * KTOT + k0 + k2];
            h2 p; p.x = (_Float16)wv.x; p.y = (_Float16)wv.y;
            *(h2*)&As[buf][m * SROW + k2] = p;
        }
        // B: im2col(x)[k0+k2..k2+1][n0+n] -> Bs[buf][n][k2..k2+1]
        // consecutive tid -> consecutive n (coalesced gathers)
        #pragma unroll
        for (int i = 0; i < (NT * KB) / (256 * 2); ++i) {
            const int idx = tid + i * 256;
            const int n  = idx & (NT - 1);
            const int k2 = (idx >> 7) << 1;
            const int ng = n0 + n;
            const int b  = ng / HW;
            const int rr = ng - b * HW;
            const int h  = rr / 56;
            const int w  = rr - h * 56;
            const int xbase = b * CIN * HW + h * 56 + w;   // x[b, 0, h, w]
            h2 p;
            #pragma unroll
            for (int j = 0; j < 2; ++j) {
                const int kg  = k0 + k2 + j;
                const int ci  = kg / 9;
                const int khw = kg - ci * 9;
                const int kh  = khw / 3;
                const int kw  = khw - kh * 3;
                const int ih  = h + kh - 1;
                const int iw  = w + kw - 1;
                float v = 0.0f;
                if ((unsigned)ih < 56u && (unsigned)iw < 56u)
                    v = X[xbase + ci * HW + (kh - 1) * 56 + (kw - 1)];
                p[j] = (_Float16)v;
            }
            *(h2*)&Bs[buf][n * SROW + k2] = p;
        }
    };

    v8f acc[4][2] = {};                 // 64 Cout x 32 pixels per wave

    stage(0, 0);
    __syncthreads();

    int cur = 0;
    for (int ks = 0; ks < NSTAGE; ++ks) {
        // overlap: issue global loads + LDS stores for next stage into other buffer
        if (ks + 1 < NSTAGE)
            stage(cur ^ 1, ks + 1);

        // compute on current buffer: 2 k-slices of 32, 8 WMMAs each
        #pragma unroll
        for (int kk = 0; kk < 2; ++kk) {
            v16h amat[4];
            #pragma unroll
            for (int mi = 0; mi < 4; ++mi) {
                const _Float16* ap = &As[cur][(wm * 64 + mi * 16 + l16) * SROW + kk * 32];
                union { v16h v; v8h h[2]; } u;
                u.h[0] = *(const v8h*)(ap + half * 8);        // K =      8*half .. +7
                u.h[1] = *(const v8h*)(ap + 16 + half * 8);   // K = 16 + 8*half .. +7
                amat[mi] = u.v;
            }
            v16h bmat[2];
            #pragma unroll
            for (int ni = 0; ni < 2; ++ni) {
                const _Float16* bp = &Bs[cur][(wn * 32 + ni * 16 + l16) * SROW + kk * 32 + half * 8];
                union { v16h v; v8h h[2]; } u;
                u.h[0] = *(const v8h*)(bp);                   // K =      8*half .. +7
                u.h[1] = *(const v8h*)(bp + 16);              // K = 16 + 8*half .. +7
                bmat[ni] = u.v;
            }
            #pragma unroll
            for (int mi = 0; mi < 4; ++mi)
                #pragma unroll
                for (int ni = 0; ni < 2; ++ni)
                    acc[mi][ni] = __builtin_amdgcn_wmma_f32_16x16x32_f16(
                        false, amat[mi], false, bmat[ni],
                        (short)0, acc[mi][ni], false, false);
        }

        __syncthreads();   // staging of next buffer done; reads of cur done
        cur ^= 1;
    }

    // ---- epilogue: bias add + store (C layout: VGPR r -> M = half*8 + r) ----
    #pragma unroll
    for (int ni = 0; ni < 2; ++ni) {
        const int ng = n0 + wn * 32 + ni * 16 + l16;
        const int b  = ng / HW;
        const int rr = ng - b * HW;
        const int h  = rr / 56;
        const int w  = rr - h * 56;
        float* obase = out + ((size_t)b * COUT * HW) + (size_t)h * 56 + w;
        #pragma unroll
        for (int mi = 0; mi < 4; ++mi) {
            const int cobase = m0 + wm * 64 + mi * 16 + half * 8;
            const v8f c = acc[mi][ni];
            #pragma unroll
            for (int r = 0; r < 8; ++r) {
                const int co = cobase + r;
                obase[(size_t)co * HW] = c[r] + bias[co];
            }
        }
    }
}

extern "C" void kernel_launch(void* const* d_in, const int* in_sizes, int n_in,
                              void* d_out, int out_size, void* d_ws, size_t ws_size,
                              hipStream_t stream) {
    const float* x    = (const float*)d_in[0];
    const float* wts  = (const float*)d_in[1];
    const float* bias = (const float*)d_in[2];
    float* out = (float*)d_out;

    dim3 grid(NTOT / NT, COUT / MT, 1);   // 784 x 2
    dim3 block(256, 1, 1);                // 8 wave32s
    conv3x3_wmma_f16<<<grid, block, 0, stream>>>(x, wts, bias, out);
}
